// DNL_38800734552408
// MI455X (gfx1250) — matrix operand order repaired and checked by reading, not compiled, not run on hardware
//
#include <hip/hip_runtime.h>
#include <hip/hip_bf16.h>
#include <math.h>

// ---------------------------------------------------------------------------
// DNL block for MI455X (gfx1250): all GEMMs via V_WMMA_F32_16X16X32_BF16.
// Pipeline:
//   k_xmean   : per-(b,c) spatial mean of x
//   k_center  : xTc[b,n,c] = bf16(x[b,c,n] - xmean)   (centering commutes w/ 1x1 conv)
//   k_wprep   : bf16 weight copies + wcv repack [tap][o][c]
//   k_small   : vmu[b,o] = wv@xmean ; maskbias[b] = bmask + wmask@xmean
//   k_proj    : WMMA GEMMs -> Qc,Kc [b,n,c] bf16 (centered), Vcn [b,c,n] bf16
//   k_masklogit/k_masksm/k_gc : SE branch (tiny)
//   k_flash   : flash attention (QK^T, online softmax, PV) + gamma/gc/residual -> yT[b,n,c]
//   k_conv    : implicit-GEMM 3x3 conv + BN + SiLU -> d_out [b,c,h,w] fp32
// ---------------------------------------------------------------------------

typedef __bf16 bf16_t;
typedef __attribute__((ext_vector_type(16))) __bf16 v16bf;
typedef __attribute__((ext_vector_type(8)))  __bf16 v8bf;
typedef __attribute__((ext_vector_type(8)))  float   v8f;

#define BDIM 8
#define CDIM 256
#define NDIM 4096
#define HDIM 64
#define WDIM 64
#define INV_SCALE_TEMP 1.25f   /* 1/(sqrt(256)*0.05) */
#define BN_EPS 1e-5f

__device__ __forceinline__ v8f wmma_bf16(v16bf a, v16bf b, v8f c) {
  // 8 args: (neg_a, A, neg_b, B, c_mod, C, reuse_a, reuse_b)
  return __builtin_amdgcn_wmma_f32_16x16x32_bf16(false, a, false, b, (short)0, c, false, false);
}

// A fragment (16x32, MxK): lane holds row m = lane&15.
// element e -> K = 16*half + e.
__device__ __forceinline__ v16bf load_a_frag(const bf16_t* base, int ld, int lane) {
  const int row = lane & 15, half = lane >> 4;
  const bf16_t* p = base + (size_t)row * ld + half * 16;
  v8bf lo = *(const v8bf*)(p);
  v8bf hi = *(const v8bf*)(p + 8);
  v16bf r;
#pragma unroll
  for (int i = 0; i < 8; ++i) { r[i] = lo[i]; r[i + 8] = hi[i]; }
  return r;
}

// B fragment (32x16, KxN): lane holds col n = lane&15, element e -> K = 16*half + e.
// Source must be contiguous along K for a fixed column (stride `ld` between cols).
__device__ __forceinline__ v16bf load_b_frag(const bf16_t* base, size_t ld, int lane) {
  const int col = lane & 15, half = lane >> 4;
  return *(const v16bf*)(base + (size_t)col * ld + half * 16);
}

// ------------------------------- prep kernels -------------------------------

__global__ __launch_bounds__(256) void k_xmean(const float* __restrict__ x,
                                               float* __restrict__ xmean) {
  __shared__ float sm[256];
  const int bc = blockIdx.x;
  const float* p = x + (size_t)bc * NDIM;
  float s = 0.f;
  for (int i = threadIdx.x; i < NDIM; i += 256) s += p[i];
  sm[threadIdx.x] = s; __syncthreads();
  for (int st = 128; st > 0; st >>= 1) {
    if (threadIdx.x < st) sm[threadIdx.x] += sm[threadIdx.x + st];
    __syncthreads();
  }
  if (threadIdx.x == 0) xmean[bc] = sm[0] * (1.0f / NDIM);
}

__global__ __launch_bounds__(256) void k_center(const float* __restrict__ x,
                                                const float* __restrict__ xmean,
                                                bf16_t* __restrict__ xTc) {
  const size_t i = (size_t)blockIdx.x * 256 + threadIdx.x;   // B*C*N threads
  const int n = (int)(i & 4095);
  const int c = (int)((i >> 12) & 255);
  const int b = (int)(i >> 20);
  xTc[((size_t)b * NDIM + n) * CDIM + c] = (bf16_t)(x[i] - xmean[b * CDIM + c]);
}

__global__ __launch_bounds__(256) void k_wprep(const float* __restrict__ wq,
                                               const float* __restrict__ wk,
                                               const float* __restrict__ wv,
                                               const float* __restrict__ wcv,
                                               bf16_t* __restrict__ wqb,
                                               bf16_t* __restrict__ wkb,
                                               bf16_t* __restrict__ wvb,
                                               bf16_t* __restrict__ wcvb) {
  const int i = blockIdx.x * 256 + threadIdx.x;
  if (i < CDIM * CDIM) {
    wqb[i] = (bf16_t)wq[i];
    wkb[i] = (bf16_t)wk[i];
    wvb[i] = (bf16_t)wv[i];
  }
  if (i < 9 * CDIM * CDIM) {
    const int tap = i % 9;
    const int c = (i / 9) % CDIM;
    const int o = i / (9 * CDIM);
    wcvb[(size_t)tap * CDIM * CDIM + (size_t)o * CDIM + c] = (bf16_t)wcv[i];
  }
}

__global__ __launch_bounds__(256) void k_small(const float* __restrict__ wv,
                                               const float* __restrict__ wmask,
                                               const float* __restrict__ bmask,
                                               const float* __restrict__ xmean,
                                               float* __restrict__ vmu,
                                               float* __restrict__ maskbias) {
  const int t = blockIdx.x * 256 + threadIdx.x;   // B*C threads
  const int b = t >> 8, o = t & 255;
  const float* xm = xmean + b * CDIM;
  float s = 0.f;
  for (int c = 0; c < CDIM; ++c) s += wv[o * CDIM + c] * xm[c];
  vmu[t] = s;
  if (o == 0) {
    float m = bmask[0];
    for (int c = 0; c < CDIM; ++c) m += wmask[c] * xm[c];
    maskbias[b] = m;
  }
}

// ------------------------------ projection GEMMs ----------------------------
// blockIdx.z: 0 -> Qc, 1 -> Kc (out[n,o] = xc^T wq^T, stored [b,n,c]),
//             2 -> Vcn (out[o,n] = wv xc + vmu, stored [b,c,n])
__global__ __launch_bounds__(32) void k_proj(const bf16_t* __restrict__ xTc,
                                             const bf16_t* __restrict__ wqb,
                                             const bf16_t* __restrict__ wkb,
                                             const bf16_t* __restrict__ wvb,
                                             const float* __restrict__ vmu,
                                             bf16_t* __restrict__ Qc,
                                             bf16_t* __restrict__ Kc,
                                             bf16_t* __restrict__ Vcn) {
  const int lane = threadIdx.x & 31;
  const int col = lane & 15, half = lane >> 4;
  const int t0 = blockIdx.x & 255;   // n-tile 0..255
  const int t1 = blockIdx.x >> 8;    // o-tile 0..15
  const int b = blockIdx.y;
  const int which = blockIdx.z;
  const int n0 = t0 * 16, o0 = t1 * 16;
  const bf16_t* w = (which == 0) ? wqb : ((which == 1) ? wkb : wvb);
  const bf16_t* xb = xTc + ((size_t)b * NDIM + n0) * CDIM;
  v8f acc = {};
  if (which < 2) {
#pragma unroll
    for (int kc = 0; kc < 8; ++kc) {
      v16bf a  = load_a_frag(xb + kc * 32, CDIM, lane);                 // rows = n
      v16bf bb = load_b_frag(w + (size_t)o0 * CDIM + kc * 32, CDIM, lane); // cols = o
      acc = wmma_bf16(a, bb, acc);
    }
    bf16_t* out = (which == 0) ? Qc : Kc;
#pragma unroll
    for (int r = 0; r < 8; ++r) {
      const int n = n0 + r + 8 * half;
      out[((size_t)b * NDIM + n) * CDIM + o0 + col] = (bf16_t)acc[r];
    }
  } else {
#pragma unroll
    for (int kc = 0; kc < 8; ++kc) {
      v16bf a  = load_a_frag(w + (size_t)o0 * CDIM + kc * 32, CDIM, lane); // rows = o
      v16bf bb = load_b_frag(xb + kc * 32, CDIM, lane);                    // cols = n
      acc = wmma_bf16(a, bb, acc);
    }
#pragma unroll
    for (int r = 0; r < 8; ++r) {
      const int o = o0 + r + 8 * half;
      Vcn[((size_t)b * CDIM + o) * NDIM + n0 + col] = (bf16_t)(acc[r] + vmu[b * CDIM + o]);
    }
  }
}

// ------------------------------- mask (SE) branch ---------------------------

__global__ __launch_bounds__(256) void k_masklogit(const bf16_t* __restrict__ xTc,
                                                   const float* __restrict__ wmask,
                                                   const float* __restrict__ maskbias,
                                                   float* __restrict__ logit) {
  const int t = blockIdx.x * 256 + threadIdx.x;  // B*N threads
  const int b = t >> 12, n = t & 4095;
  const bf16_t* p = xTc + ((size_t)b * NDIM + n) * CDIM;
  float s = maskbias[b];
  for (int c = 0; c < CDIM; ++c) s += (float)p[c] * wmask[c];
  logit[t] = s;
}

__global__ __launch_bounds__(256) void k_masksm(const float* __restrict__ logit,
                                                float* __restrict__ prob) {
  __shared__ float sm[256];
  const int b = blockIdx.x;
  const float* L = logit + (size_t)b * NDIM;
  float m = -1e30f;
  for (int i = threadIdx.x; i < NDIM; i += 256) m = fmaxf(m, L[i]);
  sm[threadIdx.x] = m; __syncthreads();
  for (int st = 128; st > 0; st >>= 1) {
    if (threadIdx.x < st) sm[threadIdx.x] = fmaxf(sm[threadIdx.x], sm[threadIdx.x + st]);
    __syncthreads();
  }
  const float M = sm[0]; __syncthreads();
  float s = 0.f;
  for (int i = threadIdx.x; i < NDIM; i += 256) s += __expf(L[i] - M);
  sm[threadIdx.x] = s; __syncthreads();
  for (int st = 128; st > 0; st >>= 1) {
    if (threadIdx.x < st) sm[threadIdx.x] += sm[threadIdx.x + st];
    __syncthreads();
  }
  const float invS = 1.f / sm[0];
  for (int i = threadIdx.x; i < NDIM; i += 256)
    prob[(size_t)b * NDIM + i] = __expf(L[i] - M) * invS;
}

__global__ __launch_bounds__(256) void k_gc(const bf16_t* __restrict__ Vcn,
                                            const float* __restrict__ prob,
                                            float* __restrict__ gc) {
  __shared__ float sm[256];
  const int bc = blockIdx.x;
  const int b = bc >> 8;
  const bf16_t* v = Vcn + (size_t)bc * NDIM;
  const float* pr = prob + (size_t)b * NDIM;
  float s = 0.f;
  for (int i = threadIdx.x; i < NDIM; i += 256) s += (float)v[i] * pr[i];
  sm[threadIdx.x] = s; __syncthreads();
  for (int st = 128; st > 0; st >>= 1) {
    if (threadIdx.x < st) sm[threadIdx.x] += sm[threadIdx.x + st];
    __syncthreads();
  }
  if (threadIdx.x == 0) gc[bc] = sm[0];
}

// -------------------------------- flash attention ---------------------------
// One wave per (b, 16-query tile). Q (16x256) and O (16x256 f32) in registers.
__global__ __launch_bounds__(32) void k_flash(const bf16_t* __restrict__ Qc,
                                              const bf16_t* __restrict__ Kc,
                                              const bf16_t* __restrict__ Vcn,
                                              const float* __restrict__ out_gc,
                                              const float* __restrict__ x,
                                              const float* __restrict__ gamma,
                                              bf16_t* __restrict__ yT) {
  __shared__ __align__(32) bf16_t Pld[16 * 32];   // wave-private P staging tile
  const int lane = threadIdx.x & 31;
  const int col = lane & 15, half = lane >> 4;
  const int b = blockIdx.y;
  const int n0 = blockIdx.x * 16;

  // loop-invariant Q A-fragments (rows = queries, K = channels)
  v16bf qf[8];
  const bf16_t* qbase = Qc + ((size_t)b * NDIM + n0) * CDIM;
#pragma unroll
  for (int kc = 0; kc < 8; ++kc) qf[kc] = load_a_frag(qbase + kc * 32, CDIM, lane);

  v8f O[16];
#pragma unroll
  for (int t = 0; t < 16; ++t) O[t] = (v8f){};
  float rowmax[8], rowsum[8];
#pragma unroll
  for (int r = 0; r < 8; ++r) { rowmax[r] = -1e30f; rowsum[r] = 0.f; }

  for (int kb = 0; kb < NDIM / 32; ++kb) {
    const int m0 = kb * 32;
    const bf16_t* k0 = Kc + ((size_t)b * NDIM + m0) * CDIM;
    if (kb + 1 < NDIM / 32)
      __builtin_prefetch(Kc + ((size_t)b * NDIM + m0 + 32) * CDIM, 0, 0);

    v8f S0 = {}, S1 = {};
#pragma unroll
    for (int kc = 0; kc < 8; ++kc) {
      S0 = wmma_bf16(qf[kc], load_b_frag(k0 + kc * 32, CDIM, lane), S0);
      S1 = wmma_bf16(qf[kc], load_b_frag(k0 + (size_t)16 * CDIM + kc * 32, CDIM, lane), S1);
    }

    float corr[8];
#pragma unroll
    for (int r = 0; r < 8; ++r) {
      const float s0 = S0[r] * INV_SCALE_TEMP;
      const float s1 = S1[r] * INV_SCALE_TEMP;
      float m = fmaxf(s0, s1);
#pragma unroll
      for (int off = 1; off < 16; off <<= 1) m = fmaxf(m, __shfl_xor(m, off, 32));
      const float nm = fmaxf(rowmax[r], m);
      corr[r] = __expf(rowmax[r] - nm);
      rowmax[r] = nm;
      const float p0 = __expf(s0 - nm), p1 = __expf(s1 - nm);
      float ps = p0 + p1;
#pragma unroll
      for (int off = 1; off < 16; off <<= 1) ps += __shfl_xor(ps, off, 32);
      rowsum[r] = rowsum[r] * corr[r] + ps;
      const int row = r + 8 * half;
      Pld[row * 32 + col]      = (bf16_t)p0;   // keys m0..m0+15
      Pld[row * 32 + 16 + col] = (bf16_t)p1;   // keys m0+16..m0+31
    }
#pragma unroll
    for (int t = 0; t < 16; ++t)
#pragma unroll
      for (int r = 0; r < 8; ++r) O[t][r] *= corr[r];

    v16bf pa = load_a_frag(Pld, 32, lane);     // transpose C-layout -> A-layout via LDS
#pragma unroll
    for (int ct = 0; ct < 16; ++ct) {
      const bf16_t* vb = Vcn + ((size_t)b * CDIM + ct * 16) * NDIM + m0;
      O[ct] = wmma_bf16(pa, load_b_frag(vb, NDIM, lane), O[ct]);
    }
  }

  const float g = gamma[0];
  float inv[8];
#pragma unroll
  for (int r = 0; r < 8; ++r) inv[r] = 1.f / rowsum[r];
#pragma unroll
  for (int ct = 0; ct < 16; ++ct) {
    const int c = ct * 16 + col;
    const float gc = out_gc[b * CDIM + c];
#pragma unroll
    for (int r = 0; r < 8; ++r) {
      const int n = n0 + r + 8 * half;
      const float xv = x[((size_t)b * CDIM + c) * NDIM + n];
      const float y = g * O[ct][r] * inv[r] + gc + xv;
      yT[((size_t)b * NDIM + n) * CDIM + c] = (bf16_t)y;
    }
  }
}

// -------------------------- 3x3 conv (implicit GEMM) + BN + SiLU ------------
__global__ __launch_bounds__(32) void k_conv(const bf16_t* __restrict__ yT,
                                             const bf16_t* __restrict__ wcvb,
                                             const float* __restrict__ bng,
                                             const float* __restrict__ bnb,
                                             const float* __restrict__ bnm,
                                             const float* __restrict__ bnv,
                                             float* __restrict__ out) {
  const int lane = threadIdx.x & 31;
  const int col = lane & 15, half = lane >> 4;
  const int pt = blockIdx.x;   // position tile 0..255 (16 contiguous w in one row)
  const int ot = blockIdx.y;   // output-channel tile 0..15
  const int b  = blockIdx.z;
  const int p0 = pt * 16;
  const int h  = p0 >> 6;
  const int w0 = p0 & 63;
  v8f acc = {};
#pragma unroll
  for (int dh = -1; dh <= 1; ++dh) {
    const int hh = h + dh;
    if (hh < 0 || hh >= HDIM) continue;
#pragma unroll
    for (int dw = -1; dw <= 1; ++dw) {
      const int tap = (dh + 1) * 3 + (dw + 1);
      const int ww = w0 + col + dw;
      const bool valid = (ww >= 0) && (ww < WDIM);
      const int wwc = valid ? ww : 0;
      const bf16_t* ybase = yT + ((size_t)b * NDIM + (size_t)hh * WDIM + wwc) * CDIM;
      const bf16_t* wbase = wcvb + (size_t)tap * CDIM * CDIM + (size_t)ot * 16 * CDIM;
#pragma unroll
      for (int kc = 0; kc < 8; ++kc) {
        v16bf a = load_a_frag(wbase + kc * 32, CDIM, lane);
        v16bf bb = {};
        if (valid) bb = *(const v16bf*)(ybase + kc * 32 + half * 16);
        acc = wmma_bf16(a, bb, acc);
      }
    }
  }
#pragma unroll
  for (int r = 0; r < 8; ++r) {
    const int o = ot * 16 + r + 8 * half;
    const float iv = bng[o] * rsqrtf(bnv[o] + BN_EPS);
    const float sh = bnb[o] - bnm[o] * iv;
    float z = acc[r] * iv + sh;
    z = z / (1.f + __expf(-z));                 // SiLU
    out[((size_t)b * CDIM + o) * NDIM + p0 + col] = z;
  }
}

// ------------------------------------ launch --------------------------------

extern "C" void kernel_launch(void* const* d_in, const int* in_sizes, int n_in,
                              void* d_out, int out_size, void* d_ws, size_t ws_size,
                              hipStream_t stream) {
  (void)in_sizes; (void)n_in; (void)out_size; (void)ws_size;
  const float* x     = (const float*)d_in[0];
  const float* wq    = (const float*)d_in[1];
  // d_in[2] (bq), d_in[4] (bk) cancel under spatial mean-centering
  const float* wk    = (const float*)d_in[3];
  const float* wv    = (const float*)d_in[5];
  const float* wmask = (const float*)d_in[6];
  const float* bmask = (const float*)d_in[7];
  const float* gamma = (const float*)d_in[8];
  const float* wcv   = (const float*)d_in[9];
  const float* bng   = (const float*)d_in[10];
  const float* bnb   = (const float*)d_in[11];
  const float* bnm   = (const float*)d_in[12];
  const float* bnv   = (const float*)d_in[13];

  char* ws = (char*)d_ws;
  bf16_t* xTc   = (bf16_t*)(ws + 0);          // 16 MB  [B][N][C]
  bf16_t* Qc    = (bf16_t*)(ws + 16777216);   // 16 MB  [B][N][C]
  bf16_t* Kc    = (bf16_t*)(ws + 33554432);   // 16 MB  [B][N][C]
  bf16_t* Vcn   = (bf16_t*)(ws + 50331648);   // 16 MB  [B][C][N]
  bf16_t* yT    = (bf16_t*)(ws + 67108864);   // 16 MB  [B][N][C]
  bf16_t* wqb   = (bf16_t*)(ws + 83886080);
  bf16_t* wkb   = (bf16_t*)(ws + 84017152);
  bf16_t* wvb   = (bf16_t*)(ws + 84148224);
  bf16_t* wcvb  = (bf16_t*)(ws + 84279296);   // [9][C][C]
  float*  xmean = (float*)(ws + 85458944);    // [B][C]
  float*  vmu   = (float*)(ws + 85467136);    // [B][C]
  float*  maskb = (float*)(ws + 85475328);    // [B]
  float*  logit = (float*)(ws + 85475584);    // [B][N]
  float*  prob  = (float*)(ws + 85606656);    // [B][N]
  float*  gc    = (float*)(ws + 85737728);    // [B][C]

  k_xmean    <<<BDIM * CDIM, 256, 0, stream>>>(x, xmean);
  k_center   <<<(BDIM * CDIM * NDIM) / 256, 256, 0, stream>>>(x, xmean, xTc);
  k_wprep    <<<(9 * CDIM * CDIM + 255) / 256, 256, 0, stream>>>(wq, wk, wv, wcv, wqb, wkb, wvb, wcvb);
  k_small    <<<BDIM, 256, 0, stream>>>(wv, wmask, bmask, xmean, vmu, maskb);
  k_proj     <<<dim3(4096, BDIM, 3), 32, 0, stream>>>(xTc, wqb, wkb, wvb, vmu, Qc, Kc, Vcn);
  k_masklogit<<<(BDIM * NDIM) / 256, 256, 0, stream>>>(xTc, wmask, maskb, logit);
  k_masksm   <<<BDIM, 256, 0, stream>>>(logit, prob);
  k_gc       <<<BDIM * CDIM, 256, 0, stream>>>(Vcn, prob, gc);
  k_flash    <<<dim3(NDIM / 16, BDIM), 32, 0, stream>>>(Qc, Kc, Vcn, gc, x, gamma, yT);
  k_conv     <<<dim3(NDIM / 16, CDIM / 16, BDIM), 32, 0, stream>>>(yT, wcvb, bng, bnb, bnm, bnv, (float*)d_out);
}
